// CvtNodeInitializer_19739669692495
// MI455X (gfx1250) — compile-verified, matrix operand order checked by reading, not run on hardware
//
#include <hip/hip_runtime.h>
#include <hip/hip_bf16.h>
#include <math.h>

#define HID 256
#define K2  512
#define NEG_INF_F (-1e30f)

typedef __attribute__((ext_vector_type(16))) __bf16 v16bf;
typedef __attribute__((ext_vector_type(8)))  __bf16 v8bf;
typedef __attribute__((ext_vector_type(8)))  float  v8f;

union AU { v16bf v; v8bf h[2]; };

// 8x f32 -> 8x bf16 (RNE) via packed v_cvt_pk_bf16_f32
__device__ __forceinline__ v8bf cvt8(float4 a, float4 b) {
    v8f f = {a.x, a.y, a.z, a.w, b.x, b.y, b.z, b.w};
    return __builtin_convertvector(f, v8bf);
}

// order-preserving float <-> uint key (for atomicMax segment-max)
__device__ __forceinline__ unsigned fkey(float f) {
    unsigned u = __float_as_uint(f);
    return (u & 0x80000000u) ? ~u : (u | 0x80000000u);
}
__device__ __forceinline__ float funkey(unsigned k) {
    return __uint_as_float((k & 0x80000000u) ? (k & 0x7FFFFFFFu) : ~k);
}

__global__ void init_out_kernel(const float* __restrict__ node,
                                const unsigned char* __restrict__ cvt,
                                const float* __restrict__ shared_cvt,
                                float* __restrict__ out, int Nn) {
    long long i = (long long)blockIdx.x * blockDim.x + threadIdx.x;
    long long total = (long long)Nn * HID;
    if (i >= total) return;
    int n = (int)(i >> 8);
    int h = (int)(i & 255);
    out[i] = cvt[n] ? shared_cvt[h] : node[i];
}

__global__ void init_stats_kernel(unsigned* __restrict__ mkey,
                                  float* __restrict__ denom, int Nn) {
    int n = blockIdx.x * blockDim.x + threadIdx.x;
    if (n >= Nn) return;
    mkey[n]  = fkey(NEG_INF_F);
    denom[n] = 0.0f;
}

// msg = [rel | node] @ W^T via v_wmma_f32_16x16x32_bf16; fused attn-dot,
// masked logits, segment-max atomicMax. Block = 256 thr (8 waves), each
// block computes 128 rows x 256 cols. Full bf16 B (512x256 = 256 KB)
// pre-swizzled into WMMA B-fragment order in LDS. B reads are software-
// pipelined 4 deep; next K-step A loads are issued before the WMMA burst.
__global__ void __launch_bounds__(256)
gemm_msg_kernel(const float* __restrict__ rel, const float* __restrict__ nod,
                const float* __restrict__ W,   const float* __restrict__ attn,
                const long long* __restrict__ tails,
                const unsigned char* __restrict__ cvt,
                float* __restrict__ msg, float* __restrict__ logits,
                unsigned* __restrict__ mkey, int E) {
    extern __shared__ char smem_raw[];
    unsigned short* Bs = (unsigned short*)smem_raw;            // 256 KB
    float* attnS = (float*)(smem_raw + 256 * 1024);            // 1 KB

    const int tid = threadIdx.x;

    // ---- stage B = W^T (bf16) into LDS in WMMA B-fragment order ----
    // fragment fr = kb*16 + nt; lane holds 16 halves (32B):
    //   h<8 : k = kb*32 + (lane>>4)*8 + h
    //   h>=8: k = kb*32 + 16 + (lane>>4)*8 + (h-8)
    //   n = nt*16 + (lane&15);  B[k][n] = W[n][k]
    for (int p = tid; p < 16 * 16 * 32; p += 256) {
        int lane = p & 31;
        int nt   = (p >> 5) & 15;
        int kb   = p >> 9;
        int n    = nt * 16 + (lane & 15);
        int k0   = kb * 32 + (lane >> 4) * 8;
        const float* w0 = W + (long long)n * K2 + k0;
        float4 a0 = ((const float4*)w0)[0];
        float4 a1 = ((const float4*)w0)[1];
        float4 b0 = ((const float4*)(w0 + 16))[0];
        float4 b1 = ((const float4*)(w0 + 16))[1];
        AU t;
        t.h[0] = cvt8(a0, a1);
        t.h[1] = cvt8(b0, b1);
        *(v16bf*)(Bs + (long long)p * 16) = t.v;
    }
    if (tid < HID) attnS[tid] = attn[tid];
    __syncthreads();

    const int wave  = tid >> 5;
    const int lane  = tid & 31;
    const int lrow  = lane & 15;   // M (A) / N (B,C) within tile
    const int lhalf = lane >> 4;
    const int mBase = blockIdx.x * 128 + wave * 16;

    int rowA = mBase + lrow;                 // A row this lane feeds
    if (rowA > E - 1) rowA = E - 1;          // clamp (stores are guarded)
    const long long rowOff = (long long)rowA * HID;

    v8f acc[16];
    v8f zero = {0.f, 0.f, 0.f, 0.f, 0.f, 0.f, 0.f, 0.f};
    #pragma unroll
    for (int nt = 0; nt < 16; ++nt) acc[nt] = zero;

    // A raw loads for one K step: two runs of 8 consecutive floats
    float4 r0, r1, r2, r3;
    {
        const int ka = 0 + lhalf * 8;
        const int kb = 16 + lhalf * 8;
        const float* s0 = rel + rowOff + ka;   // K0 = 0 -> always rel
        const float* s1 = rel + rowOff + kb;
        r0 = ((const float4*)s0)[0]; r1 = ((const float4*)s0)[1];
        r2 = ((const float4*)s1)[0]; r3 = ((const float4*)s1)[1];
    }

    #pragma unroll 1
    for (int kk = 0; kk < 16; ++kk) {        // K = 512 in steps of 32
        AU A;
        A.h[0] = cvt8(r0, r1);
        A.h[1] = cvt8(r2, r3);

        // prefetch next K-step's A while this step's WMMAs run
        if (kk + 1 < 16) {
            const int K0 = (kk + 1) * 32;
            const int ka = K0 + lhalf * 8;
            const int kb = K0 + 16 + lhalf * 8;
            const float* s0 = (ka < HID) ? (rel + rowOff + ka)
                                         : (nod + rowOff + (ka - HID));
            const float* s1 = (kb < HID) ? (rel + rowOff + kb)
                                         : (nod + rowOff + (kb - HID));
            r0 = ((const float4*)s0)[0]; r1 = ((const float4*)s0)[1];
            r2 = ((const float4*)s1)[0]; r3 = ((const float4*)s1)[1];
        }

        // B fragments, 4-deep software pipeline over the 16 N-tiles
        const unsigned short* bbase = Bs + ((long long)kk * 512 + lane) * 16;
        v16bf bq[4];
        #pragma unroll
        for (int i = 0; i < 4; ++i)
            bq[i] = *(const v16bf*)(bbase + (long long)i * 512);
        #pragma unroll
        for (int nt = 0; nt < 16; ++nt) {
            v16bf bcur = bq[nt & 3];
            if (nt + 4 < 16)
                bq[nt & 3] = *(const v16bf*)(bbase + (long long)(nt + 4) * 512);
            acc[nt] = __builtin_amdgcn_wmma_f32_16x16x32_bf16(
                false, A.v, false, bcur, (short)0, acc[nt], false, false);
        }
    }

    // ---- fused attn dot; reduce across the 16 lanes of each half ----
    float dot[8] = {0.f, 0.f, 0.f, 0.f, 0.f, 0.f, 0.f, 0.f};
    #pragma unroll
    for (int nt = 0; nt < 16; ++nt) {
        float a = attnS[nt * 16 + lrow];
        #pragma unroll
        for (int r = 0; r < 8; ++r) dot[r] += acc[nt][r] * a;
    }
    #pragma unroll
    for (int r = 0; r < 8; ++r) {
        #pragma unroll
        for (int off = 1; off < 16; off <<= 1)
            dot[r] += __shfl_xor(dot[r], off, 32);
    }

    // ---- write msg, masked logits, inline segment-max ----
    #pragma unroll
    for (int r = 0; r < 8; ++r) {
        int row = mBase + r + lhalf * 8;     // D row (C/D layout)
        if (row < E) {
            long long mo = (long long)row * HID + lrow;
            #pragma unroll
            for (int nt = 0; nt < 16; ++nt) msg[mo + nt * 16] = acc[nt][r];
            if (lrow == 0) {
                long long t = tails[row];
                bool msk = cvt[t] != 0;
                float lg = msk ? dot[r] : NEG_INF_F;
                logits[row] = lg;
                if (msk) atomicMax(&mkey[t], fkey(lg));
            }
        }
    }
}

__global__ void p_denom_kernel(const float* __restrict__ logits,
                               const long long* __restrict__ tails,
                               const unsigned* __restrict__ mkey,
                               float* __restrict__ denom,
                               float* __restrict__ pbuf, int E) {
    int e = blockIdx.x * blockDim.x + threadIdx.x;
    if (e >= E) return;
    float lg = logits[e];
    float p = 0.0f;
    if (lg > -1e29f) {
        long long t = tails[e];
        float m = funkey(mkey[t]);
        p = __expf(lg - m);
        atomicAdd(&denom[t], p);
    }
    pbuf[e] = p;
}

__global__ void __launch_bounds__(256)
scatter_kernel(const float* __restrict__ msg, const float* __restrict__ pbuf,
               const float* __restrict__ denom,
               const long long* __restrict__ tails,
               float* __restrict__ out, int E) {
    int e = blockIdx.x;
    float p = pbuf[e];
    if (p == 0.0f) return;
    long long t = tails[e];
    float d = denom[t];
    float w = p / (d > 0.0f ? d : 1.0f);
    int h = threadIdx.x;
    atomicAdd(&out[t * HID + h], msg[(long long)e * HID + h] * w);
}

extern "C" void kernel_launch(void* const* d_in, const int* in_sizes, int n_in,
                              void* d_out, int out_size, void* d_ws, size_t ws_size,
                              hipStream_t stream) {
    const float*         node       = (const float*)d_in[0];
    const float*         rel        = (const float*)d_in[1];
    const long long*     eidx       = (const long long*)d_in[2];
    const unsigned char* cvt        = (const unsigned char*)d_in[3];
    const float*         shared_cvt = (const float*)d_in[4];
    const float*         attn       = (const float*)d_in[5];
    const float*         W          = (const float*)d_in[6];
    float*               out        = (float*)d_out;

    const int Nn = in_sizes[0] / HID;
    const int E  = in_sizes[1] / HID;
    const long long* tails = eidx + E;   // edge_index[1]

    // workspace layout
    char* ws = (char*)d_ws;
    float*    msg    = (float*)ws;    ws += (size_t)E * HID * sizeof(float);
    float*    logits = (float*)ws;    ws += (size_t)E * sizeof(float);
    float*    pbuf   = (float*)ws;    ws += (size_t)E * sizeof(float);
    unsigned* mkey   = (unsigned*)ws; ws += (size_t)Nn * sizeof(unsigned);
    float*    denom  = (float*)ws;

    {
        long long total = (long long)Nn * HID;
        int blocks = (int)((total + 255) / 256);
        init_out_kernel<<<blocks, 256, 0, stream>>>(node, cvt, shared_cvt, out, Nn);
    }
    init_stats_kernel<<<(Nn + 255) / 256, 256, 0, stream>>>(mkey, denom, Nn);

    const int mBlocks = (E + 127) / 128;
    const size_t shmem = 256 * 1024 + HID * sizeof(float);  // 257 KB < 320 KB/WGP
    gemm_msg_kernel<<<mBlocks, 256, shmem, stream>>>(
        rel, node, W, attn, tails, cvt, msg, logits, mkey, E);

    p_denom_kernel<<<(E + 255) / 256, 256, 0, stream>>>(
        logits, tails, mkey, denom, pbuf, E);

    scatter_kernel<<<E, 256, 0, stream>>>(msg, pbuf, denom, tails, out, E);
}
